// T5Attention_4114578670380
// MI455X (gfx1250) — matrix operand order repaired and checked by reading, not compiled, not run on hardware
//
#include <hip/hip_runtime.h>
#include <math.h>

// ---------------------------------------------------------------------------
// T5 self-attention for MI455X (gfx1250, wave32, WMMA, async global->LDS).
// B=2, S=2048, d_model=1024, H=16, d_kv=64.  ~69 GFLOP, working set < 64MB
// (L2-resident at 192MB) -> matrix-unit bound.  f16 storage, f32 WMMA accum.
// Data movement: GLOBAL_LOAD_ASYNC_TO_LDS_B128 double-buffered staging of the
// shared operands (B tiles / K / V), consumed via ds_load_b128 WMMA gathers.
// An ILP fence keeps all 4 B tiles live at once so the ds loads clause
// together (single s_wait_dscnt) and the 4 WMMAs issue back-to-back.
// ---------------------------------------------------------------------------

#define D_MODEL 1024
#define N_HEADS 16
#define D_KV    64
#define SEQ     2048
#define BATCH   2
#define ROWS    (BATCH * SEQ)      // 4096 token rows
#define INNER   (N_HEADS * D_KV)   // 1024

typedef __attribute__((ext_vector_type(16))) _Float16 v16h;
typedef __attribute__((ext_vector_type(8)))  _Float16 v8h;
typedef __attribute__((ext_vector_type(4)))  _Float16 v4h;
typedef __attribute__((ext_vector_type(8)))  float    v8f;
typedef __attribute__((ext_vector_type(4)))  float    v4f;

static __device__ __forceinline__ v8f wmma_f32_f16(v16h a, v16h b, v8f c) {
  // 8 args: (neg_a, A, neg_b, B, c_mod, C, reuse_a, reuse_b)
  return __builtin_amdgcn_wmma_f32_16x16x32_f16(false, a, false, b, (short)0, c,
                                                false, false);
}

// ILP fence: force all four operand tiles to be materialized in distinct
// VGPR blocks before any consumer issues -> ds loads clause, WMMAs burst.
static __device__ __forceinline__ void hold4(v16h& a, v16h& b, v16h& c, v16h& d) {
  asm volatile("" : "+v"(a), "+v"(b), "+v"(c), "+v"(d));
}

// A-operand gather (16x32 MxK) from row-major [M][K] storage (ld multiple of 8,
// 16B-aligned base). ISA 7.12.2: lane m=lane&15 holds row m; lanes 0-15 hold
// K halves {0..7,16..23}; lanes 16-31 hold {8..15,24..31}. Two b128 loads.
static __device__ __forceinline__ v16h load_a(const _Float16* __restrict__ base,
                                              int ld, int lane) {
  const int m    = lane & 15;
  const int koff = (lane & 16) >> 1;          // 0 or 8
  const _Float16* p = base + m * ld + koff;
  v8h lo = *(const v8h*)(p);
  v8h hi = *(const v8h*)(p + 16);
  return __builtin_shufflevector(lo, hi, 0,1,2,3,4,5,6,7,8,9,10,11,12,13,14,15);
}

// B-operand gather (32x16 KxN) from row-major [N][K] storage (i.e. B^T stored).
// Lane n=lane&15 holds column n; lanes 0-15 hold K {0..7,16..23}, lanes 16-31
// hold K {8..15,24..31} (matches load_a). Two b128 loads (global or ds).
static __device__ __forceinline__ v16h load_b(const _Float16* __restrict__ base,
                                              int ld, int lane) {
  const int n    = lane & 15;
  const int koff = (lane & 16) >> 1;          // 0 or 8
  const _Float16* p = base + n * ld + koff;
  v8h lo = *(const v8h*)(p);
  v8h hi = *(const v8h*)(p + 16);
  return __builtin_shufflevector(lo, hi, 0,1,2,3,4,5,6,7,8,9,10,11,12,13,14,15);
}

// Asynchronously copy an R x C (halves, C multiple of 8) tile from global
// (row stride ldSrc) into LDS (packed row-major, ld = C) with
// GLOBAL_LOAD_ASYNC_TO_LDS_B128 (GV mode: per-lane LDS byte address in VDST
// VGPR, 64-bit global address in VADDR pair). Tracked by ASYNCcnt.
template <int R, int C>
static __device__ __forceinline__ void stage_tile_async(
    _Float16* dstLds, const _Float16* __restrict__ src, int ldSrc, int tid) {
  constexpr int SEGS = C / 8;                 // 16B units per row
  constexpr int PER  = (R * SEGS) / 256;      // units per thread (block = 256)
#pragma unroll
  for (int u = 0; u < PER; ++u) {
    const int unit = tid + u * 256;
    const int row  = unit / SEGS;
    const int seg  = unit % SEGS;
    const _Float16* g = src + (size_t)row * ldSrc + seg * 8;
    const unsigned  l = (unsigned)(size_t)(dstLds + row * C + seg * 8);
    asm volatile("global_load_async_to_lds_b128 %0, %1, off"
                 :: "v"(l), "v"(g)
                 : "memory");
  }
}

static __device__ __forceinline__ void wait_async0() {
  asm volatile("s_wait_asynccnt 0" ::: "memory");
}

// ---------------------------------------------------------------------------
// Kernel 1a: fp32 -> f16 elementwise cast (hidden states), 4 elems/thread.
// ---------------------------------------------------------------------------
__global__ void __launch_bounds__(256)
cast_f16_kernel(const float* __restrict__ in, _Float16* __restrict__ out, int nTot) {
  int i = (blockIdx.x * 256 + threadIdx.x) * 4;
  if (i < nTot) {
    v4f v = *(const v4f*)(in + i);
    v4h o;
#pragma unroll
    for (int j = 0; j < 4; ++j) o[j] = (_Float16)v[j];
    *(v4h*)(out + i) = o;
  }
}

// ---------------------------------------------------------------------------
// Kernel 1b: 1024x1024 fp32 [in][out] -> f16 [out][in] (LDS-tiled transpose)
// so WMMA B operands load as contiguous b128s.
// ---------------------------------------------------------------------------
__global__ void __launch_bounds__(256)
transpose_cast_kernel(const float* __restrict__ in, _Float16* __restrict__ out) {
  __shared__ float tile[32][33];
  const int bx = blockIdx.x * 32, by = blockIdx.y * 32;
  const int tx = threadIdx.x & 31, ty = threadIdx.x >> 5;  // 32x8
#pragma unroll
  for (int j = 0; j < 32; j += 8)
    tile[ty + j][tx] = in[(size_t)(by + ty + j) * D_MODEL + bx + tx];
  __syncthreads();
#pragma unroll
  for (int j = 0; j < 32; j += 8)
    out[(size_t)(bx + ty + j) * D_MODEL + by + tx] = (_Float16)tile[tx][ty + j];
}

// ---------------------------------------------------------------------------
// Kernel 1c: T5 relative-position bias expanded by distance:
// biasrel[h][(k-q)+SEQ]; hoists the log/bucket math out of the attn loop.
// ---------------------------------------------------------------------------
__global__ void __launch_bounds__(256)
bias_precompute_kernel(const float* __restrict__ table, float* __restrict__ biasrel) {
  const int idx = blockIdx.x * 256 + threadIdx.x;    // 16 * 4096
  const int h   = idx >> 12;
  const int rel = (idx & 4095) - SEQ;                // memory - context
  const int rb  = (rel > 0) ? 16 : 0;                // bidirectional half
  const int ar  = (rel < 0) ? -rel : rel;
  int bucket;
  if (ar < 8) {
    bucket = rb + ar;
  } else {
    int lg = 8 + (int)(__logf((float)ar * 0.125f) * (8.0f / __logf(16.0f)));
    bucket = rb + (lg < 15 ? lg : 15);
  }
  biasrel[idx] = table[bucket * N_HEADS + h];
}

// ---------------------------------------------------------------------------
// Kernel 2: fused QKV projections. grid=(64, 8, 3), block=256 (8 waves).
// Workgroup tile 64(M) x 128(N); wave computes 16x64 (4 C tiles).
// B tiles staged async into double-buffered LDS (shared by the 4 M-waves);
// A tile double-buffered in registers. Q,K written [B,H,S,64] f16; V written
// transposed [B,H,64,S] f16 so attention B-gathers are contiguous.
// ---------------------------------------------------------------------------
__global__ void __launch_bounds__(256)
qkv_gemm_kernel(const _Float16* __restrict__ X,
                const _Float16* __restrict__ WqT, const _Float16* __restrict__ WkT,
                const _Float16* __restrict__ WvT,
                _Float16* __restrict__ Qo, _Float16* __restrict__ Ko,
                _Float16* __restrict__ Vto) {
  __shared__ __align__(16) _Float16 LdsB[2][128 * 32];   // 16KB double buffer
  const int tid   = threadIdx.x;
  const int lane  = tid & 31;
  const int wave  = tid >> 5;
  const int which = blockIdx.z;
  const _Float16* __restrict__ Wt = (which == 0) ? WqT : (which == 1) ? WkT : WvT;
  const int mbase  = blockIdx.x * 64 + (wave & 3) * 16;
  const int nblock = blockIdx.y * 128;
  const int nbase  = nblock + (wave >> 2) * 64;

  const _Float16* __restrict__ Xrow = X + (size_t)mbase * D_MODEL;
  const _Float16* __restrict__ Wrow = Wt + (size_t)nblock * D_MODEL;

  stage_tile_async<128, 32>(LdsB[0], Wrow, D_MODEL, tid);
  v16h a_cur = load_a(Xrow, D_MODEL, lane);

  v8f acc[4] = {};
  for (int kc = 0; kc < D_MODEL; kc += 32) {
    const int cur = (kc >> 5) & 1;
    wait_async0();                // this wave's stage of LdsB[cur] landed
    __syncthreads();              // ...and everyone else's; prev readers done
    v16h a_nxt = a_cur;
    if (kc + 32 < D_MODEL) {      // overlap next-chunk staging with WMMAs
      stage_tile_async<128, 32>(LdsB[cur ^ 1], Wrow + kc + 32, D_MODEL, tid);
      a_nxt = load_a(Xrow + kc + 32, D_MODEL, lane);
    }
    const _Float16* Bl = LdsB[cur] + ((wave >> 2) * 64) * 32;
    // load all 4 B tiles first, fence them live, then WMMA burst
    v16h bm0 = load_b(Bl +  0 * 32, 32, lane);
    v16h bm1 = load_b(Bl + 16 * 32, 32, lane);
    v16h bm2 = load_b(Bl + 32 * 32, 32, lane);
    v16h bm3 = load_b(Bl + 48 * 32, 32, lane);
    hold4(bm0, bm1, bm2, bm3);
    acc[0] = wmma_f32_f16(a_cur, bm0, acc[0]);
    acc[1] = wmma_f32_f16(a_cur, bm1, acc[1]);
    acc[2] = wmma_f32_f16(a_cur, bm2, acc[2]);
    acc[3] = wmma_f32_f16(a_cur, bm3, acc[3]);
    a_cur = a_nxt;
  }

  const int n    = lane & 15;
  const int roff = (lane & 16) >> 1;          // C/D: lanes 16-31 hold rows +8
#pragma unroll
  for (int t = 0; t < 4; ++t) {
#pragma unroll
    for (int r = 0; r < 8; ++r) {
      const int M  = mbase + roff + r;
      const int N  = nbase + t * 16 + n;
      const int bb = M >> 11, s = M & (SEQ - 1);
      const int hh = N >> 6,  d = N & (D_KV - 1);
      const int bh = bb * N_HEADS + hh;
      const _Float16 v = (_Float16)acc[t][r];
      if (which == 0)      Qo [((size_t)bh * SEQ + s) * D_KV + d] = v;
      else if (which == 1) Ko [((size_t)bh * SEQ + s) * D_KV + d] = v;
      else                 Vto[((size_t)bh * D_KV + d) * SEQ + s] = v;
    }
  }
}

// ---------------------------------------------------------------------------
// Kernel 3: flash attention. grid=(S/128, B*H), block=256 (8 waves).
// K and V^T chunks (32 keys) staged async into double-buffered LDS shared by
// all 8 waves. Each wave owns a 16-row q tile: scores (4 WMMA) -> +bias ->
// online softmax (shfl reductions, v_exp_f32) -> P through per-wave LDS
// (C-layout f16 stores, A-layout b128 reloads) -> context (4 WMMA).
// ---------------------------------------------------------------------------
__global__ void __launch_bounds__(256)
attn_kernel(const _Float16* __restrict__ Q, const _Float16* __restrict__ K,
            const _Float16* __restrict__ Vt, const float* __restrict__ biasrel,
            _Float16* __restrict__ ctx) {
  __shared__ __align__(16) _Float16 LdsK[2][32 * 64];   // 8KB
  __shared__ __align__(16) _Float16 LdsV[2][64 * 32];   // 8KB
  __shared__ __align__(16) _Float16 Pbuf[8][16 * 32];   // 8KB, 1KB/wave

  const int tid   = threadIdx.x;
  const int lane  = tid & 31;
  const int wave  = tid >> 5;
  const int bh    = blockIdx.y;
  const int h     = bh & (N_HEADS - 1);
  const int b     = bh >> 4;
  const int qbase = blockIdx.x * 128 + wave * 16;

  const _Float16* __restrict__ Qh = Q  + (size_t)bh * SEQ * D_KV;
  const _Float16* __restrict__ Kh = K  + (size_t)bh * SEQ * D_KV;
  const _Float16* __restrict__ Vh = Vt + (size_t)bh * D_KV * SEQ;
  const float*    __restrict__ bt = biasrel + (size_t)h * (2 * SEQ);
  _Float16* P = Pbuf[wave];

  stage_tile_async<32, 64>(LdsK[0], Kh, D_KV, tid);   // keys 0..31, all 64 d
  stage_tile_async<64, 32>(LdsV[0], Vh, SEQ, tid);    // all 64 d, keys 0..31

  // Q tile resident in registers: A-operand halves over d=0..31, 32..63
  const v16h qa0 = load_a(Qh + (size_t)qbase * D_KV +  0, D_KV, lane);
  const v16h qa1 = load_a(Qh + (size_t)qbase * D_KV + 32, D_KV, lane);

  v8f acc0 = {}, acc1 = {}, acc2 = {}, acc3 = {};
  float mrun[8], lrun[8];
#pragma unroll
  for (int r = 0; r < 8; ++r) { mrun[r] = -1e30f; lrun[r] = 0.0f; }

  const int n    = lane & 15;
  const int roff = (lane & 16) >> 1;

  for (int kb = 0; kb < SEQ; kb += 32) {
    const int cur = (kb >> 5) & 1;
    wait_async0();
    __syncthreads();
    if (kb + 32 < SEQ) {          // overlap next chunk staging with compute
      stage_tile_async<32, 64>(LdsK[cur ^ 1], Kh + (size_t)(kb + 32) * D_KV,
                               D_KV, tid);
      stage_tile_async<64, 32>(LdsV[cur ^ 1], Vh + (kb + 32), SEQ, tid);
    }
    const _Float16* Kc = LdsK[cur];
    const _Float16* Vc = LdsV[cur];

    // ---- scores: S[16x32] = Q[16x64] @ K^T[64x32] (keys kb..kb+31) ----
    v16h kb00 = load_b(Kc +  0 * 64 +  0, 64, lane);
    v16h kb01 = load_b(Kc +  0 * 64 + 32, 64, lane);
    v16h kb10 = load_b(Kc + 16 * 64 +  0, 64, lane);
    v16h kb11 = load_b(Kc + 16 * 64 + 32, 64, lane);
    hold4(kb00, kb01, kb10, kb11);
    v8f s0 = {}, s1 = {};
    s0 = wmma_f32_f16(qa0, kb00, s0);
    s0 = wmma_f32_f16(qa1, kb01, s0);
    s1 = wmma_f32_f16(qa0, kb10, s1);
    s1 = wmma_f32_f16(qa1, kb11, s1);

    // ---- bias add + online softmax update (rows live per 16-lane half) ----
    const int k0 = kb + n;
    const int k1 = kb + 16 + n;
#pragma unroll
    for (int r = 0; r < 8; ++r) {
      const int q = qbase + roff + r;
      float x0 = s0[r] + bt[k0 - q + SEQ];
      float x1 = s1[r] + bt[k1 - q + SEQ];
      float mx = fmaxf(x0, x1);
#pragma unroll
      for (int d = 1; d < 16; d <<= 1) mx = fmaxf(mx, __shfl_xor(mx, d, 32));
      const float mnew  = fmaxf(mrun[r], mx);
      const float scale = __expf(mrun[r] - mnew);
      const float p0 = __expf(x0 - mnew);
      const float p1 = __expf(x1 - mnew);
      float rsum = p0 + p1;
#pragma unroll
      for (int d = 1; d < 16; d <<= 1) rsum += __shfl_xor(rsum, d, 32);
      lrun[r] = lrun[r] * scale + rsum;
      mrun[r] = mnew;
      acc0[r] *= scale; acc1[r] *= scale; acc2[r] *= scale; acc3[r] *= scale;
      P[(roff + r) * 32 + n]      = (_Float16)p0;   // row-major 16x32 in LDS
      P[(roff + r) * 32 + 16 + n] = (_Float16)p1;
    }

    // ---- context: acc[16x64] += P[16x32] @ V[32x64] ----
    const v16h pa  = load_a(P, 32, lane);            // ds_load_b128 x2
    v16h vb0 = load_b(Vc +  0 * 32, 32, lane);
    v16h vb1 = load_b(Vc + 16 * 32, 32, lane);
    v16h vb2 = load_b(Vc + 32 * 32, 32, lane);
    v16h vb3 = load_b(Vc + 48 * 32, 32, lane);
    hold4(vb0, vb1, vb2, vb3);
    acc0 = wmma_f32_f16(pa, vb0, acc0);
    acc1 = wmma_f32_f16(pa, vb1, acc1);
    acc2 = wmma_f32_f16(pa, vb2, acc2);
    acc3 = wmma_f32_f16(pa, vb3, acc3);
  }

  // ---- normalize + store ctx [B,S,INNER] (heads re-interleaved) ----
#pragma unroll
  for (int r = 0; r < 8; ++r) {
    const float inv = 1.0f / lrun[r];
    _Float16* dst = ctx + (size_t)(b * SEQ + qbase + roff + r) * INNER + h * D_KV;
    dst[ 0 + n] = (_Float16)(acc0[r] * inv);
    dst[16 + n] = (_Float16)(acc1[r] * inv);
    dst[32 + n] = (_Float16)(acc2[r] * inv);
    dst[48 + n] = (_Float16)(acc3[r] * inv);
  }
}

// ---------------------------------------------------------------------------
// Kernel 4: output projection ctx[4096x1024]f16 @ w_o -> fp32 d_out.
// Same async-LDS double-buffered scheme as the QKV GEMM.
// ---------------------------------------------------------------------------
__global__ void __launch_bounds__(256)
out_proj_kernel(const _Float16* __restrict__ Ctx, const _Float16* __restrict__ WoT,
                float* __restrict__ out) {
  __shared__ __align__(16) _Float16 LdsB[2][128 * 32];
  const int tid   = threadIdx.x;
  const int lane  = tid & 31;
  const int wave  = tid >> 5;
  const int mbase  = blockIdx.x * 64 + (wave & 3) * 16;
  const int nblock = blockIdx.y * 128;
  const int nbase  = nblock + (wave >> 2) * 64;

  const _Float16* __restrict__ Xrow = Ctx + (size_t)mbase * INNER;
  const _Float16* __restrict__ Wrow = WoT + (size_t)nblock * INNER;

  stage_tile_async<128, 32>(LdsB[0], Wrow, INNER, tid);
  v16h a_cur = load_a(Xrow, INNER, lane);

  v8f acc[4] = {};
  for (int kc = 0; kc < INNER; kc += 32) {
    const int cur = (kc >> 5) & 1;
    wait_async0();
    __syncthreads();
    v16h a_nxt = a_cur;
    if (kc + 32 < INNER) {
      stage_tile_async<128, 32>(LdsB[cur ^ 1], Wrow + kc + 32, INNER, tid);
      a_nxt = load_a(Xrow + kc + 32, INNER, lane);
    }
    const _Float16* Bl = LdsB[cur] + ((wave >> 2) * 64) * 32;
    v16h bm0 = load_b(Bl +  0 * 32, 32, lane);
    v16h bm1 = load_b(Bl + 16 * 32, 32, lane);
    v16h bm2 = load_b(Bl + 32 * 32, 32, lane);
    v16h bm3 = load_b(Bl + 48 * 32, 32, lane);
    hold4(bm0, bm1, bm2, bm3);
    acc[0] = wmma_f32_f16(a_cur, bm0, acc[0]);
    acc[1] = wmma_f32_f16(a_cur, bm1, acc[1]);
    acc[2] = wmma_f32_f16(a_cur, bm2, acc[2]);
    acc[3] = wmma_f32_f16(a_cur, bm3, acc[3]);
    a_cur = a_nxt;
  }
  const int n    = lane & 15;
  const int roff = (lane & 16) >> 1;
#pragma unroll
  for (int t = 0; t < 4; ++t)
#pragma unroll
    for (int r = 0; r < 8; ++r)
      out[(size_t)(mbase + roff + r) * D_MODEL + nbase + t * 16 + n] = acc[t][r];
}

// ---------------------------------------------------------------------------
// Host-side launch. Workspace layout (bytes):
//   [0,8M)       hidden16 (f16)  -- later reused as ctx16 (attn runs after qkv)
//   [8M,16M)     wqT,wkT,wvT,woT (f16, [out][in], 2MB each)
//   [16M,24M)    Q  f16 [B,H,S,64]
//   [24M,32M)    K  f16 [B,H,S,64]
//   [32M,40M)    V^T f16 [B,H,64,S]
//   [40M,40.25M) biasrel fp32 [H][4096]
// ---------------------------------------------------------------------------
extern "C" void kernel_launch(void* const* d_in, const int* in_sizes, int n_in,
                              void* d_out, int out_size, void* d_ws, size_t ws_size,
                              hipStream_t stream) {
  (void)in_sizes; (void)n_in; (void)out_size; (void)ws_size;
  const float* hidden = (const float*)d_in[0];
  const float* w_q    = (const float*)d_in[1];
  const float* w_k    = (const float*)d_in[2];
  const float* w_v    = (const float*)d_in[3];
  const float* w_o    = (const float*)d_in[4];
  const float* reltab = (const float*)d_in[5];

  char* ws = (char*)d_ws;
  const size_t SZ_HID = (size_t)ROWS * D_MODEL * sizeof(_Float16);    // 8 MB
  const size_t SZ_W   = (size_t)D_MODEL * D_MODEL * sizeof(_Float16); // 2 MB
  const size_t SZ_QKV = (size_t)BATCH * N_HEADS * SEQ * D_KV * sizeof(_Float16); // 8 MB

  _Float16* hidden16 = (_Float16*)(ws);
  _Float16* ctx16    = hidden16;                 // alias: hidden dead after QKV
  _Float16* wqT      = (_Float16*)(ws + SZ_HID);
  _Float16* wkT      = (_Float16*)(ws + SZ_HID + 1 * SZ_W);
  _Float16* wvT      = (_Float16*)(ws + SZ_HID + 2 * SZ_W);
  _Float16* woT      = (_Float16*)(ws + SZ_HID + 3 * SZ_W);
  _Float16* q16      = (_Float16*)(ws + SZ_HID + 4 * SZ_W);
  _Float16* k16      = (_Float16*)(ws + SZ_HID + 4 * SZ_W + 1 * SZ_QKV);
  _Float16* vt16     = (_Float16*)(ws + SZ_HID + 4 * SZ_W + 2 * SZ_QKV);
  float*    biasrel  = (float*)   (ws + SZ_HID + 4 * SZ_W + 3 * SZ_QKV);

  // 1) precision conversion + weight transposes + bias table
  cast_f16_kernel<<<(ROWS * D_MODEL) / (256 * 4), 256, 0, stream>>>(
      hidden, hidden16, ROWS * D_MODEL);
  transpose_cast_kernel<<<dim3(32, 32), 256, 0, stream>>>(w_q, wqT);
  transpose_cast_kernel<<<dim3(32, 32), 256, 0, stream>>>(w_k, wkT);
  transpose_cast_kernel<<<dim3(32, 32), 256, 0, stream>>>(w_v, wvT);
  transpose_cast_kernel<<<dim3(32, 32), 256, 0, stream>>>(w_o, woT);
  bias_precompute_kernel<<<(N_HEADS * 2 * SEQ) / 256, 256, 0, stream>>>(
      reltab, biasrel);

  // 2) QKV projections (WMMA + async-LDS B staging)
  qkv_gemm_kernel<<<dim3(ROWS / 64, D_MODEL / 128, 3), 256, 0, stream>>>(
      hidden16, wqT, wkT, wvT, q16, k16, vt16);

  // 3) flash attention (WMMA + async-LDS K/V staging + online softmax)
  attn_kernel<<<dim3(SEQ / 128, BATCH * N_HEADS), 256, 0, stream>>>(
      q16, k16, vt16, biasrel, ctx16);

  // 4) output projection -> fp32 result
  out_proj_kernel<<<dim3(ROWS / 64, D_MODEL / 128), 256, 0, stream>>>(
      ctx16, woT, (float*)d_out);
}
